// Decoder_54090818126517
// MI455X (gfx1250) — matrix operand order repaired and checked by reading, not compiled, not run on hardware
//
#include <hip/hip_runtime.h>

// ---- CDNA5 WMMA types ----
typedef __attribute__((ext_vector_type(16))) __bf16         v16bf;
typedef __attribute__((ext_vector_type(16))) unsigned short v16us;
typedef __attribute__((ext_vector_type(8)))  unsigned int   v8u;
typedef __attribute__((ext_vector_type(8)))  float          v8f;

union BFrag { v16us us; v16bf bf; v8u w; };

// Hardware transcendental tanh (CDNA5 TRANS32 op). v_nop covers the 1-op
// TRANS hazard window (ISA 7.4: 1 independent op before result use).
__device__ __forceinline__ float tanh_hw(float x) {
  float r;
  asm("v_tanh_f32 %0, %1\n\tv_nop" : "=v"(r) : "v"(x));
  return r;
}
// Packed RNE f32->bf16 convert: lo -> [15:0], hi -> [31:16].
__device__ __forceinline__ unsigned cvt_pk_bf16(float lo, float hi) {
  unsigned r;
  asm("v_cvt_pk_bf16_f32 %0, %1, %2" : "=v"(r) : "v"(lo), "v"(hi));
  return r;
}
__device__ __forceinline__ float sigm_hw(float x) {
  return 0.5f * (tanh_hw(0.5f * x) + 1.0f);
}

// ============================================================================
// Generic small-M GEMM: C[16x16 tile] = act(A[M,K] @ B[K,N] + bias)  (bf16 WMMA)
// one wave per 16x16 output tile; per-lane gather + packed f32->bf16 convert.
// A-frag: lane l, dword j: M = l&15, K pair = 2j + (j>=4?8:0) + 8*(l>>4)
// B-frag: lane l, dword j: N = l&15, K pair = 2j + (j>=4?8:0) + 8*(l>>4)
// ============================================================================
__global__ void gemm_wmma_bf16(const float* __restrict__ A, int lda,
                               const float* __restrict__ Bm, int ldb,
                               const float* __restrict__ bias,
                               float* __restrict__ C, int ldc,
                               int K, int act, int acc)
{
  const int lane = threadIdx.x & 31;
  const int hi   = lane >> 4;
  const int col  = lane & 15;
  const int n    = blockIdx.x * 16 + col;   // output column of this lane
  const int am   = blockIdx.y * 16 + col;   // A row supplied by this lane
  const int m0   = blockIdx.y * 16;

  v8f c;
  if (acc) {
#pragma unroll
    for (int r = 0; r < 8; ++r) c[r] = C[(long)(m0 + r + 8 * hi) * ldc + n];
  } else {
    float bv = bias ? bias[n] : 0.0f;
#pragma unroll
    for (int r = 0; r < 8; ++r) c[r] = bv;
  }

  const float* Arow = A + (long)am * lda;
  for (int k0 = 0; k0 < K; k0 += 32) {
    BFrag a, b;
#pragma unroll
    for (int j = 0; j < 8; ++j) {
      int ka = 2 * j + ((j >= 4) ? 8 : 0) + 8 * hi;   // k pair (ka, ka+1)
      float2 av = *(const float2*)(Arow + k0 + ka);   // 8B-aligned (ka even)
      a.w[j] = cvt_pk_bf16(av.x, av.y);
      int kb = 2 * j + ((j >= 4) ? 8 : 0) + 8 * hi;   // k pair (kb, kb+1)
      float b0v = Bm[(long)(k0 + kb) * ldb + n];
      float b1v = Bm[(long)(k0 + kb + 1) * ldb + n];
      b.w[j] = cvt_pk_bf16(b0v, b1v);
    }
    c = __builtin_amdgcn_wmma_f32_16x16x32_bf16(false, a.bf, false, b.bf,
                                                (short)0, c, false, false);
  }
  if (act == 1) {
#pragma unroll
    for (int r = 0; r < 8; ++r) c[r] = tanh_hw(c[r]);
  }
#pragma unroll
  for (int r = 0; r < 8; ++r) C[(long)(m0 + r + 8 * hi) * ldc + n] = c[r];
}

// ============================================================================
// Attention scores: scores[b,s] += sum_n tanh( (enc@W_e)[m,n] + hpre[b,n] )*v[n]
// Block: 256 thr (8 waves) = one 16-row M tile x one 512-col N half.
// LDS-staged A (16x32 bf16) and W_e^T (512x32 bf16), padded stride 34.
// ============================================================================
#define AST 34
__global__ void attn_scores_wmma(const float* __restrict__ enc,
                                 const float* __restrict__ attnW,
                                 const float* __restrict__ hpre,
                                 const float* __restrict__ vvec,
                                 float* __restrict__ scores)
{
  __shared__ unsigned short sA[16 * AST];   // [row][k]
  __shared__ unsigned short sB[512 * AST];  // [n][k] transposed W_e slice
  __shared__ float sred[16];

  const int b    = blockIdx.x >> 5;         // S/16 = 32 M-tiles per batch
  const int s0   = (blockIdx.x & 31) << 4;
  const int half = blockIdx.y;              // N columns [half*512, half*512+512)
  const int t    = threadIdx.x;
  const int lane = t & 31;
  const int wave = t >> 5;
  const int hi   = lane >> 4;
  const int col  = lane & 15;

  if (t < 16) sred[t] = 0.0f;

  const float* rowbase = enc + (long)(b * 512 + s0) * 2048;
  const float* We      = attnW + 1024 * 1024;   // rows H..3H-1 of attn_W

  v8f c[4];
#pragma unroll
  for (int j = 0; j < 4; ++j)
#pragma unroll
    for (int r = 0; r < 8; ++r) c[j][r] = 0.0f;

  const unsigned* pA = (const unsigned*)sA;
  const unsigned* pB = (const unsigned*)sB;
  unsigned*       qA = (unsigned*)sA;
  unsigned*       qB = (unsigned*)sB;

  for (int kk = 0; kk < 2048; kk += 32) {
    __syncthreads();
    // stage A: 16 rows x 32 k; thread handles one k-consecutive pair
    {
      int e  = t * 2;                 // pair base element
      int r  = e >> 5, k = e & 31;    // k even
      float2 av = *(const float2*)(rowbase + (long)r * 2048 + kk + k);
      qA[(r * AST + k) >> 1] = cvt_pk_bf16(av.x, av.y);
    }
    // stage B^T: 32 k x 512 n; pair over k rows (coalesced along n)
#pragma unroll
    for (int i = 0; i < 32; ++i) {
      int p = i * 256 + t;            // 8192 pairs
      int n = p & 511;
      int k = (p >> 9) * 2;
      const float* g = We + (long)(kk + k) * 1024 + half * 512 + n;
      qB[(n * AST + k) >> 1] = cvt_pk_bf16(g[0], g[1024]);
    }
    __syncthreads();

    BFrag a;
#pragma unroll
    for (int j = 0; j < 8; ++j) {
      int k0 = 2 * j + ((j >= 4) ? 8 : 0) + 8 * hi;     // pair (k0,k0+1)
      a.w[j] = pA[(col * AST + k0) >> 1];
    }
#pragma unroll
    for (int j4 = 0; j4 < 4; ++j4) {
      int nloc = (wave * 4 + j4) * 16 + col;
      BFrag bfr;
#pragma unroll
      for (int j = 0; j < 8; ++j) {
        int kb = 2 * j + ((j >= 4) ? 8 : 0) + 8 * hi;   // pair (kb,kb+1)
        bfr.w[j] = pB[(nloc * AST + kb) >> 1];
      }
      c[j4] = __builtin_amdgcn_wmma_f32_16x16x32_bf16(false, a.bf, false, bfr.bf,
                                                      (short)0, c[j4], false, false);
    }
  }

  // epilogue: +hpre, hardware tanh, * v[n], reduce over n
  float rs[8];
#pragma unroll
  for (int r = 0; r < 8; ++r) rs[r] = 0.0f;
#pragma unroll
  for (int j4 = 0; j4 < 4; ++j4) {
    int n = half * 512 + (wave * 4 + j4) * 16 + col;
    float hv = hpre[b * 1024 + n];
    float vv = vvec[n];
#pragma unroll
    for (int r = 0; r < 8; ++r)
      rs[r] += tanh_hw(c[j4][r] + hv) * vv;
  }
#pragma unroll
  for (int r = 0; r < 8; ++r) {
    rs[r] += __shfl_xor(rs[r], 1, 32);
    rs[r] += __shfl_xor(rs[r], 2, 32);
    rs[r] += __shfl_xor(rs[r], 4, 32);
    rs[r] += __shfl_xor(rs[r], 8, 32);   // stays within 16-lane half
    if (col == 0) atomicAdd(&sred[r + 8 * hi], rs[r]);
  }
  __syncthreads();
  if (t < 16) atomicAdd(&scores[b * 512 + s0 + t], sred[t]);
}

// ============================================================================
// softmax over S + context = attn_w @ enc   (enc is L2-resident on 2nd pass)
// ============================================================================
__global__ void softmax_context(const float* __restrict__ scores,
                                const float* __restrict__ enc,
                                float* __restrict__ context)
{
  __shared__ float w[512];
  __shared__ float red[256];
  const int b = blockIdx.x, t = threadIdx.x;
  float v0 = scores[b * 512 + t], v1 = scores[b * 512 + 256 + t];
  red[t] = fmaxf(v0, v1);
  __syncthreads();
  for (int o = 128; o > 0; o >>= 1) {
    if (t < o) red[t] = fmaxf(red[t], red[t + o]);
    __syncthreads();
  }
  float m = red[0];
  __syncthreads();
  float e0 = __expf(v0 - m), e1 = __expf(v1 - m);
  w[t] = e0; w[256 + t] = e1;
  red[t] = e0 + e1;
  __syncthreads();
  for (int o = 128; o > 0; o >>= 1) {
    if (t < o) red[t] += red[t + o];
    __syncthreads();
  }
  float inv = 1.0f / red[0];
  __syncthreads();
  w[t] *= inv; w[256 + t] *= inv;
  __syncthreads();
  const float* eb = enc + (long)b * 512 * 2048;
  for (int d = t; d < 2048; d += 256) {
    float acc = 0.0f;
    for (int s = 0; s < 512; ++s) acc += w[s] * eb[(long)s * 2048 + d];
    context[b * 2048 + d] = acc;
  }
}

// ---- elementwise / concat helpers ----
__global__ void build_rnn_in(const int* __restrict__ x, const float* __restrict__ emb,
                             const float* __restrict__ context, float* __restrict__ out)
{
  int i = blockIdx.x * blockDim.x + threadIdx.x;   // 32*3072
  int b = i / 3072, cidx = i % 3072;
  out[i] = (cidx < 1024) ? emb[(long)x[b] * 1024 + cidx]
                         : context[b * 2048 + (cidx - 1024)];
}

__global__ void build_concat(const float* __restrict__ a1024,
                             const float* __restrict__ context, float* __restrict__ out)
{
  int i = blockIdx.x * blockDim.x + threadIdx.x;   // 32*3072
  int b = i / 3072, cidx = i % 3072;
  out[i] = (cidx < 1024) ? a1024[b * 1024 + cidx]
                         : context[b * 2048 + (cidx - 1024)];
}

__global__ void lstm_cell(const float* __restrict__ gates, const float* __restrict__ c_prev,
                          float* __restrict__ h_new, float* __restrict__ c_new)
{
  int idx = blockIdx.x * blockDim.x + threadIdx.x;  // 32*1024
  int b = idx >> 10, j = idx & 1023;
  const float* g = gates + b * 4096;
  float i_ = sigm_hw(g[j]);
  float f_ = sigm_hw(g[1024 + j]);
  float gg = tanh_hw(g[2048 + j]);
  float o_ = sigm_hw(g[3072 + j]);
  float c = f_ * c_prev[idx] + i_ * gg;
  c_new[idx] = c;
  h_new[idx] = o_ * tanh_hw(c);
}

// ============================================================================
extern "C" void kernel_launch(void* const* d_in, const int* in_sizes, int n_in,
                              void* d_out, int out_size, void* d_ws, size_t ws_size,
                              hipStream_t stream) {
  const int*   x     = (const int*)  d_in[0];
  const float* h0    = (const float*)d_in[1];   // [2,32,1024]
  const float* c0    = (const float*)d_in[2];   // [2,32,1024]
  const float* enc   = (const float*)d_in[3];   // [32,512,2048]
  const float* emb   = (const float*)d_in[4];
  const float* attnW = (const float*)d_in[5];   // [3072,1024]
  const float* attnB = (const float*)d_in[6];
  const float* vvec  = (const float*)d_in[7];
  const float* Wih0  = (const float*)d_in[8];   // [3072,4096]
  const float* Whh0  = (const float*)d_in[9];   // [1024,4096]
  const float* b0    = (const float*)d_in[10];
  const float* Wih1  = (const float*)d_in[11];  // [1024,4096]
  const float* Whh1  = (const float*)d_in[12];
  const float* b1    = (const float*)d_in[13];
  const float* out1W = (const float*)d_in[14];  // [3072,1024]
  const float* out1b = (const float*)d_in[15];
  const float* out2W = (const float*)d_in[16];  // [1024,32000]
  const float* out2b = (const float*)d_in[17];

  float* ws      = (float*)d_ws;
  float* hpre    = ws;             // 32*1024
  float* scores  = ws + 32768;     // 32*512
  float* context = ws + 49152;     // 32*2048
  float* rnn_in  = ws + 114688;    // 32*3072
  float* gates0  = ws + 212992;    // 32*4096
  float* gates1  = ws + 344064;    // 32*4096
  float* hdn     = ws + 475136;    // 32*1024
  float* ovec    = ws + 507904;    // 32*3072

  float* preds = (float*)d_out;            // [32,32000]
  float* hnew0 = preds + 32 * 32000;       // h_new layer0
  float* hnew1 = hnew0 + 32 * 1024;        // h_new layer1
  float* cnew0 = hnew1 + 32 * 1024;        // c_new layer0
  float* cnew1 = cnew0 + 32 * 1024;        // c_new layer1

  // 1) hpre = h_top @ W_h + attn_b    (h_top = h0[-1] = layer 1)
  gemm_wmma_bf16<<<dim3(64, 2), 32, 0, stream>>>(h0 + 32768, 1024, attnW, 1024,
                                                 attnB, hpre, 1024, 1024, 0, 0);
  // 2) attention scores (fused energy GEMM + tanh + v-dot)
  hipMemsetAsync(scores, 0, 32 * 512 * sizeof(float), stream);
  attn_scores_wmma<<<dim3(1024, 2), 256, 0, stream>>>(enc, attnW, hpre, vvec, scores);
  // 3) softmax + context
  softmax_context<<<32, 256, 0, stream>>>(scores, enc, context);
  // 4) rnn_in = [embed(x), context]
  build_rnn_in<<<384, 256, 0, stream>>>(x, emb, context, rnn_in);
  // 5) gates0 = rnn_in @ W_ih0 + b0 ; += h0[0] @ W_hh0
  gemm_wmma_bf16<<<dim3(256, 2), 32, 0, stream>>>(rnn_in, 3072, Wih0, 4096,
                                                  b0, gates0, 4096, 3072, 0, 0);
  gemm_wmma_bf16<<<dim3(256, 2), 32, 0, stream>>>(h0, 1024, Whh0, 4096,
                                                  nullptr, gates0, 4096, 1024, 0, 1);
  // 6) LSTM cell 0
  lstm_cell<<<128, 256, 0, stream>>>(gates0, c0, hnew0, cnew0);
  // 7) gates1 = h0_new @ W_ih1 + b1 ; += h0[1] @ W_hh1
  gemm_wmma_bf16<<<dim3(256, 2), 32, 0, stream>>>(hnew0, 1024, Wih1, 4096,
                                                  b1, gates1, 4096, 1024, 0, 0);
  gemm_wmma_bf16<<<dim3(256, 2), 32, 0, stream>>>(h0 + 32768, 1024, Whh1, 4096,
                                                  nullptr, gates1, 4096, 1024, 0, 1);
  // 8) LSTM cell 1
  lstm_cell<<<128, 256, 0, stream>>>(gates1, c0 + 32768, hnew1, cnew1);
  // 9) o = [h1_new, context]
  build_concat<<<384, 256, 0, stream>>>(hnew1, context, ovec);
  // 10) hdn = tanh(o @ out1_W + out1_b)
  gemm_wmma_bf16<<<dim3(64, 2), 32, 0, stream>>>(ovec, 3072, out1W, 1024,
                                                 out1b, hdn, 1024, 3072, 1, 0);
  // 11) predictions = hdn @ out2_W + out2_b   (bandwidth-bound: 131 MB weight)
  gemm_wmma_bf16<<<dim3(2000, 2), 32, 0, stream>>>(hdn, 1024, out2W, 32000,
                                                   out2b, preds, 32000, 1024, 0, 0);
}
